// _GraphWaveNet_62577673503632
// MI455X (gfx1250) — compile-verified, matrix operand order checked by reading, not run on hardware
//
#include <hip/hip_runtime.h>

// ---------------- types for CDNA5 WMMA / TDM ----------------
typedef __attribute__((ext_vector_type(16))) __bf16 v16bf;
typedef __attribute__((ext_vector_type(8)))  __bf16 v8bf;
typedef __attribute__((ext_vector_type(8)))  float  v8f;
typedef __attribute__((ext_vector_type(4)))  unsigned int u32x4;
typedef __attribute__((ext_vector_type(8)))  int i32x8;
typedef __attribute__((ext_vector_type(4)))  int i32x4;

#define NN 2048            // nodes
#define KDIM 2048          // GEMM K == nodes
#define RC 32
#define SC 256
#define EC 512
#define NOUT 12
#define NCHUNK 7

#if __has_builtin(__builtin_amdgcn_tensor_load_to_lds)
#define HAVE_TDM 1
#else
#define HAVE_TDM 0
#endif

// ---------------- bf16 fragment loader ----------------
// 16-bit A/B fragment layout (CDNA5 ISA 7.12.2): r=lane&15 selects the row
// (A) / column (B); sel=lane>>4 selects K-halves:
//   halves[0..7]  = K = kbase + sel*8 + {0..7}
//   halves[8..15] = K = kbase + 16 + sel*8 + {0..7}
__device__ inline v16bf load_frag16(const __bf16* p) {
  union { v16bf v; v8bf h[2]; } u;
  u.h[0] = *(const v8bf*)(p);
  u.h[1] = *(const v8bf*)(p + 16);
  return u.v;
}

#define WMMA_BF16(A, B, C) \
  __builtin_amdgcn_wmma_f32_16x16x32_bf16(false, (A), false, (B), (short)0, (C), false, false)

// ---------------- WMMA GEMM: Y[m,w] = sum_v X[m,v] * AT[w,v] ----------------
// X: M x 2048 bf16 row-major, AT: 2048 x 2048 bf16 row-major (transposed
// support), Y: M x 2048 bf16 row-major, f32 accumulation. M % 64 == 0.
// Block = 256 threads = 8 waves as 2(M) x 4(N); wave tile 32x64
// (2 M-tiles x 4 N-tiles = 8 accumulators), software-pipelined fragment
// double buffer so loads for step k+1 overlap WMMAs of step k.
__global__ __launch_bounds__(256)
void k_gemm_bf16(const __bf16* __restrict__ X, const __bf16* __restrict__ AT,
                 __bf16* __restrict__ Y) {
  const int tid  = threadIdx.x;
  const int lane = tid & 31;
  const int wid  = tid >> 5;
  const int mw   = wid & 1;
  const int nw   = wid >> 1;
  const int m0   = blockIdx.y * 64 + mw * 32;
  const int n0   = blockIdx.x * 256 + nw * 64;
  const int row  = lane & 15;
  const int sel  = lane >> 4;

  const __bf16* ap = X  + (size_t)(m0 + row) * KDIM + sel * 8;
  const __bf16* bp = AT + (size_t)(n0 + row) * KDIM + sel * 8;

  v8f c00 = {}, c01 = {}, c02 = {}, c03 = {};
  v8f c10 = {}, c11 = {}, c12 = {}, c13 = {};

  v16bf a0 = load_frag16(ap);
  v16bf a1 = load_frag16(ap + 16 * KDIM);
  v16bf b0 = load_frag16(bp);
  v16bf b1 = load_frag16(bp + 16 * KDIM);
  v16bf b2 = load_frag16(bp + 32 * KDIM);
  v16bf b3 = load_frag16(bp + 48 * KDIM);

  for (int k = 32; k < KDIM; k += 32) {
    v16bf a0n = load_frag16(ap + k);
    v16bf a1n = load_frag16(ap + 16 * KDIM + k);
    v16bf b0n = load_frag16(bp + k);
    v16bf b1n = load_frag16(bp + 16 * KDIM + k);
    v16bf b2n = load_frag16(bp + 32 * KDIM + k);
    v16bf b3n = load_frag16(bp + 48 * KDIM + k);

    c00 = WMMA_BF16(a0, b0, c00);
    c01 = WMMA_BF16(a0, b1, c01);
    c02 = WMMA_BF16(a0, b2, c02);
    c03 = WMMA_BF16(a0, b3, c03);
    c10 = WMMA_BF16(a1, b0, c10);
    c11 = WMMA_BF16(a1, b1, c11);
    c12 = WMMA_BF16(a1, b2, c12);
    c13 = WMMA_BF16(a1, b3, c13);

    a0 = a0n; a1 = a1n;
    b0 = b0n; b1 = b1n; b2 = b2n; b3 = b3n;
  }
  c00 = WMMA_BF16(a0, b0, c00);
  c01 = WMMA_BF16(a0, b1, c01);
  c02 = WMMA_BF16(a0, b2, c02);
  c03 = WMMA_BF16(a0, b3, c03);
  c10 = WMMA_BF16(a1, b0, c10);
  c11 = WMMA_BF16(a1, b1, c11);
  c12 = WMMA_BF16(a1, b2, c12);
  c13 = WMMA_BF16(a1, b3, c13);

  // D layout: VGPR r -> M = mbase + sel*8 + r, N = nbase + (lane&15)
  const int ncol = n0 + row;
  {
    const int mrow = m0 + sel * 8;
#pragma unroll
    for (int r = 0; r < 8; ++r) {
      Y[(size_t)(mrow + r) * NN + ncol +  0] = (__bf16)c00[r];
      Y[(size_t)(mrow + r) * NN + ncol + 16] = (__bf16)c01[r];
      Y[(size_t)(mrow + r) * NN + ncol + 32] = (__bf16)c02[r];
      Y[(size_t)(mrow + r) * NN + ncol + 48] = (__bf16)c03[r];
    }
  }
  {
    const int mrow = m0 + 16 + sel * 8;
#pragma unroll
    for (int r = 0; r < 8; ++r) {
      Y[(size_t)(mrow + r) * NN + ncol +  0] = (__bf16)c10[r];
      Y[(size_t)(mrow + r) * NN + ncol + 16] = (__bf16)c11[r];
      Y[(size_t)(mrow + r) * NN + ncol + 32] = (__bf16)c12[r];
      Y[(size_t)(mrow + r) * NN + ncol + 48] = (__bf16)c13[r];
    }
  }
}

#if HAVE_TDM
// Build a D# (ISA §8.3/8.4) for a 2D bf16 tile load: tile_h rows of
// tile_w elements, row stride = stride_elems, huge tensor dims (no OOB).
__device__ inline void tdm_load_2d(unsigned lds_off, const void* gaddr,
                                   unsigned tile_w, unsigned tile_h,
                                   unsigned long long stride_elems) {
  const unsigned long long ga = (unsigned long long)gaddr;
  const unsigned tw = 1u << 30;   // tensor_dim0 (large, no OOB)
  const unsigned th = 1u << 20;   // tensor_dim1 (large, no OOB)
  u32x4 g0 = {
    1u,                                            // count=1 user descriptor
    lds_off,                                       // lds_addr (bytes)
    (unsigned)(ga & 0xffffffffu),                  // global_addr[31:0]
    (unsigned)((ga >> 32) & 0x01ffffffu) | 0x80000000u  // addr[56:32] | type=2
  };
  i32x8 g1 = {
    (int)(1u << 16),                               // data_size=1 (2B), mask=0
    (int)((tw & 0xffffu) << 16),                   // tensor_dim0[15:0]
    (int)(((tw >> 16) & 0xffffu) | ((th & 0xffffu) << 16)),  // dim0 hi | dim1 lo
    (int)(((th >> 16) & 0xffffu) | (tile_w << 16)),          // dim1 hi | tile_dim0
    (int)(tile_h & 0xffffu),                       // tile_dim1 (tile_dim2=0)
    (int)(unsigned)(stride_elems & 0xffffffffu),   // tensor_dim0_stride[31:0]
    (int)(unsigned)((stride_elems >> 32) & 0xffffu),  // stride[47:32]
    0
  };
  i32x4 z4 = {0, 0, 0, 0};
#if defined(__clang_major__) && (__clang_major__ >= 23)
  i32x8 z8 = {0, 0, 0, 0, 0, 0, 0, 0};
  __builtin_amdgcn_tensor_load_to_lds(g0, g1, z4, z4, z8, 0);
#else
  __builtin_amdgcn_tensor_load_to_lds(g0, g1, z4, z4, 0);
#endif
}
#endif

// ---------------- transpose + bf16 convert of A[2] supports ----------------
__global__ __launch_bounds__(256)
void k_transpose_bf16(const float* __restrict__ A, __bf16* __restrict__ AT) {
  __shared__ float tile[32][33];
  const int s = blockIdx.z;
  const float* Ai = A + (size_t)s * NN * NN;
  __bf16* ATi = AT + (size_t)s * NN * NN;
  const int bx = blockIdx.x * 32, by = blockIdx.y * 32;
  const int tx = threadIdx.x & 31, ty = threadIdx.x >> 5;
  for (int r = ty; r < 32; r += 8)
    tile[r][tx] = Ai[(size_t)(by + r) * NN + bx + tx];
  __syncthreads();
  for (int r = ty; r < 32; r += 8)
    ATi[(size_t)(bx + r) * NN + by + tx] = (__bf16)tile[tx][r];
}

// ---------------- adp = softmax(relu(nv1 @ nv2), axis=1), stored transposed bf16 ----------------
__global__ __launch_bounds__(256)
void k_adp(const float* __restrict__ nv1, const float* __restrict__ nv2,
           __bf16* __restrict__ adpT) {
  __shared__ float rowbuf[NN];
  __shared__ float red[256];
  const int v = blockIdx.x, tid = threadIdx.x;
  float a[10];
#pragma unroll
  for (int k = 0; k < 10; ++k) a[k] = nv1[v * 10 + k];
  float lmax = -1e30f;
  for (int w = tid; w < NN; w += 256) {
    float t = 0.f;
#pragma unroll
    for (int k = 0; k < 10; ++k) t += a[k] * nv2[k * NN + w];
    t = fmaxf(t, 0.f);
    rowbuf[w] = t;
    lmax = fmaxf(lmax, t);
  }
  red[tid] = lmax; __syncthreads();
  for (int s2 = 128; s2 > 0; s2 >>= 1) {
    if (tid < s2) red[tid] = fmaxf(red[tid], red[tid + s2]);
    __syncthreads();
  }
  const float mx = red[0]; __syncthreads();
  float lsum = 0.f;
  for (int w = tid; w < NN; w += 256) {
    float e = expf(rowbuf[w] - mx);
    rowbuf[w] = e;
    lsum += e;
  }
  red[tid] = lsum; __syncthreads();
  for (int s2 = 128; s2 > 0; s2 >>= 1) {
    if (tid < s2) red[tid] += red[tid + s2];
    __syncthreads();
  }
  const float inv = 1.f / red[0];
  for (int w = tid; w < NN; w += 256)
    adpT[(size_t)w * NN + v] = (__bf16)(rowbuf[w] * inv);
}

// ---------------- start conv: (pad T->13) then 2->32 1x1 ----------------
__global__ __launch_bounds__(256)
void k_start(const float* __restrict__ xin, const float* __restrict__ sw,
             const float* __restrict__ sb, float* __restrict__ X) {
  size_t idx = (size_t)blockIdx.x * 256 + threadIdx.x; // over 16*32*2048*13
  const int t = idx % 13; size_t r = idx / 13;
  const int n = r % NN; r /= NN;
  const int o = r % RC; const int b = r / RC;
  float v = sb[o];
  if (t > 0) {
    float x0 = xin[(((size_t)b * 2 + 0) * NN + n) * 12 + (t - 1)];
    float x1 = xin[(((size_t)b * 2 + 1) * NN + n) * 12 + (t - 1)];
    v += sw[o * 2 + 0] * x0 + sw[o * 2 + 1] * x1;
  }
  X[idx] = v;
}

// ---------------- dilated conv (k=2) + tanh*sigmoid gate -> bf16 chunk0 ----------------
// X layout: [b][c][n][t] (stride Tin). chunk0 layout: [(b*32+c)*Tout + t][n] bf16.
__global__ __launch_bounds__(256)
void k_dconv(const float* __restrict__ X,
             const float* __restrict__ fw, const float* __restrict__ fb,
             const float* __restrict__ gw, const float* __restrict__ gb,
             __bf16* __restrict__ chunk0, int Tin, int d) {
  const int Tout = Tin - d;
  __shared__ float xs[32 * 16 * 13];
  __shared__ float wf[32 * 32 * 2];
  __shared__ float wg[32 * 32 * 2];
  const int b = blockIdx.y, n0 = blockIdx.x * 16, tid = threadIdx.x;
  for (int i = tid; i < 2048; i += 256) { wf[i] = fw[i]; wg[i] = gw[i]; }
  const int tot = 32 * 16 * Tin;
  for (int i = tid; i < tot; i += 256) {
    const int t = i % Tin; int rr = i / Tin;
    const int nl = rr & 15; const int c = rr >> 4;
    xs[(c * 16 + nl) * Tin + t] = X[(((size_t)b * RC + c) * NN + n0 + nl) * Tin + t];
  }
  __syncthreads();
  for (int idx = tid; idx < 512; idx += 256) {
    const int nl = idx & 15, o = idx >> 4;
    for (int t = 0; t < Tout; ++t) {
      float f = fb[o], g = gb[o];
#pragma unroll
      for (int c = 0; c < 32; ++c) {
        const float xa = xs[(c * 16 + nl) * Tin + t];
        const float xb = xs[(c * 16 + nl) * Tin + t + d];
        f += wf[(o * 32 + c) * 2 + 0] * xa + wf[(o * 32 + c) * 2 + 1] * xb;
        g += wg[(o * 32 + c) * 2 + 0] * xa + wg[(o * 32 + c) * 2 + 1] * xb;
      }
      const float val = tanhf(f) * (1.f / (1.f + expf(-g)));
      chunk0[(((size_t)b * RC + o) * Tout + t) * NN + n0 + nl] = (__bf16)val;
    }
  }
}

// ---------------- skip conv: only the final time slice survives ----------------
// skip accumulator layout: [b][n][256]
__global__ __launch_bounds__(256)
void k_skip(const __bf16* __restrict__ chunk0, const float* __restrict__ sw,
            const float* __restrict__ sb, float* __restrict__ skip, int Tout) {
  __shared__ float g[32];
  const int n = blockIdx.x, b = blockIdx.y, sc = threadIdx.x;
  if (sc < 32)
    g[sc] = (float)chunk0[(((size_t)b * RC + sc) * Tout + (Tout - 1)) * NN + n];
  __syncthreads();
  float acc = sb[sc];
#pragma unroll
  for (int c = 0; c < 32; ++c) acc += sw[sc * 32 + c] * g[c];
  skip[((size_t)b * NN + n) * SC + sc] += acc;
}

// ---------------- gcn 224->32 conv + residual + batchnorm -> new X ----------------
// Tile staging via the Tensor Data Mover: 7 x (32 rows x 32 bf16) tiles,
// row stride Tout*2048 elements, landed contiguously in LDS.
__global__ __launch_bounds__(256)
void k_gcn(const __bf16* __restrict__ chunks, size_t cstride,
           const float* __restrict__ gw, const float* __restrict__ gb,
           const float* __restrict__ bng, const float* __restrict__ bnb,
           const float* __restrict__ bnm, const float* __restrict__ bnv,
           const float* __restrict__ Xres, float* __restrict__ Xout,
           int Tin, int Tout) {
  __shared__ __bf16 hb[NCHUNK * 32 * 32];   // [j][c][nl]
  const int n0 = blockIdx.x * 32, t = blockIdx.y, b = blockIdx.z, tid = threadIdx.x;
#if HAVE_TDM
  if (tid == 0) {
    const unsigned lds0 = (unsigned)(size_t)&hb[0];
    for (int j = 0; j < NCHUNK; ++j) {
      const __bf16* g =
          chunks + (size_t)j * cstride + ((size_t)b * RC * Tout + t) * NN + n0;
      tdm_load_2d(lds0 + (unsigned)j * 32 * 32 * 2, g, 32, 32,
                  (unsigned long long)Tout * NN);
    }
  }
  __builtin_amdgcn_s_wait_tensorcnt(0);
  __syncthreads();
#else
  for (int i = tid; i < NCHUNK * 32 * 32; i += 256) {
    const int nl = i & 31, c = (i >> 5) & 31, j = i >> 10;
    hb[i] = chunks[(size_t)j * cstride +
                   (((size_t)b * RC + c) * Tout + t) * NN + n0 + nl];
  }
  __syncthreads();
#endif
  const int nl = tid & 31, og0 = (tid >> 5) * 4;
  const int off = Tin - Tout;
#pragma unroll
  for (int og = 0; og < 4; ++og) {
    const int o = og0 + og;
    float acc = gb[o];
    for (int k = 0; k < NCHUNK * 32; ++k)
      acc += gw[o * 224 + k] * (float)hb[k * 32 + nl];
    const float res = Xres[(((size_t)b * RC + o) * NN + n0 + nl) * Tin + t + off];
    const float inv = bng[o] * rsqrtf(bnv[o] + 1e-5f);
    Xout[(((size_t)b * RC + o) * NN + n0 + nl) * Tout + t] =
        (acc + res - bnm[o]) * inv + bnb[o];
  }
}

// ---------------- end: relu(skip) -> 512 relu -> 12 ----------------
__global__ __launch_bounds__(256)
void k_end(const float* __restrict__ skip, const float* __restrict__ w1,
           const float* __restrict__ b1, const float* __restrict__ w2,
           const float* __restrict__ b2, float* __restrict__ out) {
  __shared__ float h1[SC];
  __shared__ float h2[EC];
  const int n = blockIdx.x, b = blockIdx.y, tid = threadIdx.x;
  h1[tid] = fmaxf(skip[((size_t)b * NN + n) * SC + tid], 0.f);
  __syncthreads();
  for (int e = tid; e < EC; e += 256) {
    float acc = b1[e];
    for (int s = 0; s < SC; ++s) acc += w1[e * SC + s] * h1[s];
    h2[e] = fmaxf(acc, 0.f);
  }
  __syncthreads();
  if (tid < NOUT) {
    float acc = b2[tid];
    for (int e = 0; e < EC; ++e) acc += w2[tid * EC + e] * h2[e];
    out[((size_t)b * NOUT + tid) * NN + n] = acc;
  }
}

// ---------------- host orchestration ----------------
extern "C" void kernel_launch(void* const* d_in, const int* in_sizes, int n_in,
                              void* d_out, int out_size, void* d_ws, size_t ws_size,
                              hipStream_t stream) {
  (void)in_sizes; (void)n_in; (void)out_size; (void)ws_size;
  const float* x        = (const float*)d_in[0];
  const float* A        = (const float*)d_in[1];
  const float* nv1      = (const float*)d_in[2];
  const float* nv2      = (const float*)d_in[3];
  const float* filter_w = (const float*)d_in[4];
  const float* filter_b = (const float*)d_in[5];
  const float* gate_w   = (const float*)d_in[6];
  const float* gate_b   = (const float*)d_in[7];
  const float* skip_w   = (const float*)d_in[8];
  const float* skip_b   = (const float*)d_in[9];
  const float* gcn_w    = (const float*)d_in[10];
  const float* gcn_b    = (const float*)d_in[11];
  const float* bn_g     = (const float*)d_in[12];
  const float* bn_b     = (const float*)d_in[13];
  const float* bn_m     = (const float*)d_in[14];
  const float* bn_v     = (const float*)d_in[15];
  const float* start_w  = (const float*)d_in[16];
  const float* start_b  = (const float*)d_in[17];
  const float* e1w      = (const float*)d_in[18];
  const float* e1b      = (const float*)d_in[19];
  const float* e2w      = (const float*)d_in[20];
  const float* e2b      = (const float*)d_in[21];

  // workspace carve-up (all 256B aligned)
  char* p = (char*)d_ws;
  const size_t szAT = (size_t)3 * NN * NN * sizeof(__bf16);      // 25.2 MB
  const size_t szX  = (size_t)16 * RC * NN * 13 * sizeof(float); // 54.5 MB
  const size_t cs   = (size_t)16 * RC * 12 * NN;                 // chunk elems (max M=6144)
  const size_t szCH = (size_t)NCHUNK * cs * sizeof(__bf16);      // 176 MB
  const size_t szSK = (size_t)16 * NN * SC * sizeof(float);      // 33.5 MB
  __bf16* ATb     = (__bf16*)p;           p += szAT;
  float*  X       = (float*)p;            p += szX;
  float*  X2      = (float*)p;            p += szX;
  __bf16* chunks  = (__bf16*)p;           p += szCH;
  float*  skipacc = (float*)p;            p += szSK;

  // one-time-per-call prep
  k_transpose_bf16<<<dim3(64, 64, 2), 256, 0, stream>>>(A, ATb);
  k_adp<<<NN, 256, 0, stream>>>(nv1, nv2, ATb + (size_t)2 * NN * NN);
  hipMemsetAsync(skipacc, 0, szSK, stream);
  {
    const size_t tot = (size_t)16 * RC * NN * 13;
    k_start<<<(unsigned)(tot / 256), 256, 0, stream>>>(x, start_w, start_b, X);
  }

  const int dil[8] = {1, 2, 1, 2, 1, 2, 1, 2};
  int Tin = 13;
  float* Xa = X; float* Xb2 = X2;
  for (int i = 0; i < 8; ++i) {
    const int d = dil[i];
    const int Tout = Tin - d;
    __bf16* c0 = chunks;

    k_dconv<<<dim3(128, 16), 256, 0, stream>>>(
        Xa, filter_w + (size_t)i * 2048, filter_b + i * 32,
        gate_w + (size_t)i * 2048, gate_b + i * 32, c0, Tin, d);

    k_skip<<<dim3(NN, 16), 256, 0, stream>>>(
        c0, skip_w + (size_t)i * 8192, skip_b + i * 256, skipacc, Tout);

    const int Mrows = 16 * RC * Tout; // 512*T, always a multiple of 64
    const dim3 ggrid(NN / 256, Mrows / 64);
    for (int s = 0; s < 3; ++s) {
      const __bf16* At = ATb + (size_t)s * NN * NN;
      __bf16* y1 = chunks + (size_t)(1 + 2 * s) * cs;
      __bf16* y2 = chunks + (size_t)(2 + 2 * s) * cs;
      k_gemm_bf16<<<ggrid, 256, 0, stream>>>(c0, At, y1);
      k_gemm_bf16<<<ggrid, 256, 0, stream>>>(y1, At, y2);
    }

    k_gcn<<<dim3(NN / 32, Tout, 16), 256, 0, stream>>>(
        chunks, cs, gcn_w + (size_t)i * 7168, gcn_b + i * 32,
        bn_g + i * 32, bn_b + i * 32, bn_m + i * 32, bn_v + i * 32,
        Xa, Xb2, Tin, Tout);

    float* tmp = Xa; Xa = Xb2; Xb2 = tmp;
    Tin = Tout;
  }

  k_end<<<dim3(NN, 16), 256, 0, stream>>>(skipacc, e1w, e1b, e2w, e2b, (float*)d_out);
}
